// ConvAttention_23905787970123
// MI455X (gfx1250) — matrix operand order repaired and checked by reading, not compiled, not run on hardware
//
#include <hip/hip_runtime.h>
#include <hip/hip_bf16.h>
#include <math.h>

typedef __attribute__((ext_vector_type(16))) __bf16 v16bf;
typedef __attribute__((ext_vector_type(8)))  __bf16 v8bf;
typedef __attribute__((ext_vector_type(8)))  float  v8f;

// Problem constants (B,C,H,W,S = 8,64,64,64,12; K=5)
#define NB 8
#define NC 64
#define NH 64
#define NW 64
#define NS 12
#define CSTRIDE (NH*NW*NS)      // 49152 floats between channels
#define BSTRIDE ((size_t)NC*CSTRIDE)

// ---------------------------------------------------------------------------
// Kernel 1: fold the K-branch (1x1 conv then 5x5 conv are both linear):
//   weff[tap][ci] = sum_co w2[64+co][tap] * w1[64+co][ci]
//   wsum[tap]     = sum_co b1[64+co] * w2[64+co][tap]   (bias field taps)
//   w1v (bf16)    = w1[128..191][:]  (V projection), b1v = b1[128..191]
// ---------------------------------------------------------------------------
__global__ __launch_bounds__(256) void precompute_kernel(
    const float* __restrict__ w1, const float* __restrict__ b1,
    const float* __restrict__ w2,
    float* __restrict__ weff, float* __restrict__ wsum,
    float* __restrict__ b1v, __bf16* __restrict__ w1v) {
  const int tid = threadIdx.x;
  for (int i = tid; i < 25 * 64; i += 256) {
    const int tap = i >> 6, ci = i & 63;
    float acc = 0.f;
    #pragma unroll 4
    for (int co = 0; co < 64; ++co)
      acc = fmaf(w2[(64 + co) * 25 + tap], w1[(64 + co) * 64 + ci], acc);
    weff[i] = acc;
  }
  if (tid < 25) {
    float acc = 0.f;
    for (int c = 0; c < 64; ++c)
      acc = fmaf(b1[64 + c], w2[(64 + c) * 25 + tid], acc);
    wsum[tid] = acc;
  }
  if (tid < 64) b1v[tid] = b1[128 + tid];
  for (int i = tid; i < 64 * 64; i += 256)
    w1v[i] = (__bf16)w1[128 * 64 + i];
}

// ---------------------------------------------------------------------------
// Kernel 2: effective 64->1 5x5 conv (LDS-tiled, 8-channel chunks) + softmax
// over t.  Block = (b, h, 16-wide w segment), 192 threads = 16 pixels x 12 t.
// Next chunk is prefetched (global_prefetch) while the VALU computes.
// ---------------------------------------------------------------------------
__global__ __launch_bounds__(192) void attn_kernel(
    const float* __restrict__ x, const float* __restrict__ weff,
    const float* __restrict__ wsum, float* __restrict__ attn) {
  __shared__ float lweff[25 * 64];
  __shared__ float lwsum[25];
  __shared__ float xs[8 * 5 * 20 * 12];   // [ci_chunk][kh][w+halo][t]
  __shared__ float sa[192];

  const int tid  = threadIdx.x;
  const int blk  = blockIdx.x;            // b*256 + h*4 + wseg
  const int wseg = blk & 3;
  const int h    = (blk >> 2) & 63;
  const int b    = blk >> 8;
  const int w0   = wseg * 16;
  const int pw   = tid / 12;
  const int t    = tid % 12;
  const int w    = w0 + pw;

  for (int i = tid; i < 25 * 64; i += 192) lweff[i] = weff[i];
  if (tid < 25) lwsum[tid] = wsum[tid];

  const float* xb = x + (size_t)b * BSTRIDE;
  const float4 zero4 = make_float4(0.f, 0.f, 0.f, 0.f);
  const int wwlo = (w0 - 2 < 0) ? 0 : w0 - 2;
  float a = 0.f;
  for (int c0 = 0; c0 < 64; c0 += 8) {
    __syncthreads();
    // Stage x[c0..c0+7, h-2..h+2, w0-2..w0+17, :] with zero-padded halo.
    // S=12 is a multiple of 4, so each float4 stays inside one (ci,hh,ww).
    for (int i4 = tid; i4 < 2400; i4 += 192) {
      const int tq = i4 % 3;            // which float4 of the 12 t's
      const int wl = (i4 / 3) % 20;
      const int dh = (i4 / 60) % 5;
      const int cl = i4 / 300;
      const int hh = h + dh - 2;
      const int ww = w0 + wl - 2;
      float4 v = zero4;
      if (hh >= 0 && hh < NH && ww >= 0 && ww < NW)
        v = *(const float4*)&xb[(size_t)(c0 + cl) * CSTRIDE +
                                (hh * NW + ww) * NS + tq * 4];
      ((float4*)xs)[i4] = v;
    }
    __syncthreads();
    // Prefetch next chunk's rows (one 128-B cacheline per slot) while we
    // compute: 8 ch x 5 rows x 8 lines = 320 slots.
    if (c0 + 8 < 64) {
      for (int i = tid; i < 320; i += 192) {
        const int line = i & 7;
        const int dh   = (i >> 3) % 5;
        const int cl   = i / 40;
        const int hh   = h + dh - 2;
        if (hh >= 0 && hh < NH)
          __builtin_prefetch(&xb[(size_t)(c0 + 8 + cl) * CSTRIDE +
                                 (hh * NW + wwlo) * NS + line * 32], 0, 3);
      }
    }
    #pragma unroll
    for (int cl = 0; cl < 8; ++cl) {
      #pragma unroll
      for (int kh = 0; kh < 5; ++kh) {
        const float* xr = &xs[((cl * 5 + kh) * 20 + pw) * 12 + t];
        const float* wr = &lweff[(kh * 5) * 64 + (c0 + cl)];
        #pragma unroll
        for (int kw = 0; kw < 5; ++kw)
          a = fmaf(wr[kw * 64], xr[kw * 12], a);
      }
    }
  }
  // Boundary-aware bias field from b1_K (SAME zero padding).
  float bias = 0.f;
  for (int kh = 0; kh < 5; ++kh) {
    const int hh = h + kh - 2;
    if (hh < 0 || hh >= NH) continue;
    for (int kw = 0; kw < 5; ++kw) {
      const int ww = w + kw - 2;
      if (ww >= 0 && ww < NW) bias += lwsum[kh * 5 + kw];
    }
  }
  a += bias;
  __syncthreads();
  sa[tid] = a;
  __syncthreads();
  const int base = pw * 12;
  float m = sa[base];
  #pragma unroll
  for (int j = 1; j < 12; ++j) m = fmaxf(m, sa[base + j]);
  float denom = 0.f;
  #pragma unroll
  for (int j = 0; j < 12; ++j) denom += __expf(sa[base + j] - m);
  attn[((size_t)b * (NH * NW) + h * NW + w) * NS + t] = __expf(a - m) / denom;
}

// ---------------------------------------------------------------------------
// Kernel 3: y[p][ci] = sum_t attn[p][t]*x[ci][p][t]  (bf16, TRANSPOSED in
// LDS so B fragments are contiguous), then out = w1v(64x64) @ y + b1v via
// v_wmma_f32_16x16x32_bf16, replicated over the 12 s slots.
// Block = (b, 64-pixel tile), 256 threads = 8 waves; each wave owns one
// M-block (A fragments loaded once) and two N-subtiles.
// ---------------------------------------------------------------------------
#define LYP 72   // ly row stride in bf16 (144 B, 16-B aligned)
__global__ __launch_bounds__(256) void gemm_out_kernel(
    const float* __restrict__ x, const float* __restrict__ attn,
    const __bf16* __restrict__ w1v, const float* __restrict__ b1v,
    float* __restrict__ out) {
  __shared__ float  lat[64 * 12];       // attn tile
  __shared__ __bf16 ly[64 * LYP];       // [pixel][ci]  (K along rows)
  __shared__ __bf16 lw[64 * 64];        // [co][ci]
  __shared__ float  lb[64];

  const int tid  = threadIdx.x;
  const int b    = blockIdx.x >> 6;
  const int tile = blockIdx.x & 63;
  const int p0   = tile * 64;

  for (int i = tid; i < 192; i += 256)
    ((float4*)lat)[i] =
        ((const float4*)(attn + ((size_t)b * (NH * NW) + p0) * NS))[i];
  for (int i = tid; i < 2048; i += 256)      // dword copy of 64x64 bf16
    ((unsigned int*)lw)[i] = ((const unsigned int*)w1v)[i];
  if (tid < 64) lb[tid] = b1v[tid];
  __syncthreads();

  // Time reduction: y = sum_t attn * x  (coalesced 48-B aligned reads)
  const float* xb = x + (size_t)b * BSTRIDE;
  for (int i = tid; i < 4096; i += 256) {
    const int ci = i >> 6;
    const int p  = i & 63;
    const float4* xp = (const float4*)(xb + (size_t)ci * CSTRIDE +
                                       (size_t)(p0 + p) * NS);
    const float4* ap = (const float4*)&lat[p * 12];
    float acc = 0.f;
    #pragma unroll
    for (int q = 0; q < 3; ++q) {
      const float4 xv = xp[q], av = ap[q];
      acc = fmaf(av.x, xv.x, acc);
      acc = fmaf(av.y, xv.y, acc);
      acc = fmaf(av.z, xv.z, acc);
      acc = fmaf(av.w, xv.w, acc);
    }
    ly[p * LYP + ci] = (__bf16)acc;     // transposed store
  }
  __syncthreads();

  const int lane = tid & 31;
  const int wv   = tid >> 5;
  const int g    = lane >> 4;   // half-wave group
  const int ln   = lane & 15;
  const int mi   = wv >> 1;     // M block: same for both of this wave's tiles

  // A (16x32 bf16) fragments, one per K-step: lane ln, half g -> two
  // contiguous 8-element runs: K = kbase+8g..+7 (v0-3) and +16 (v4-7).
  v16bf afk[2];
  #pragma unroll
  for (int kb = 0; kb < 2; ++kb) {
    const v8bf* pa =
        (const v8bf*)&lw[(mi * 16 + ln) * 64 + kb * 32 + 8 * g];
    const v8bf a_lo = pa[0];
    const v8bf a_hi = pa[2];            // +16 elements
    afk[kb] = __builtin_shufflevector(
        a_lo, a_hi, 0, 1, 2, 3, 4, 5, 6, 7, 8, 9, 10, 11, 12, 13, 14, 15);
  }

  #pragma unroll
  for (int sti = 0; sti < 2; ++sti) {
    const int ni = (wv * 2 + sti) & 3;  // N block (pixel)
    v8f acc = {};
    #pragma unroll
    for (int kb = 0; kb < 2; ++kb) {
      // B (32x16 bf16): lane n=ln, half g -> two contiguous 8-element runs:
      //   K = kbase+8g..+7 (v0-3) and +16 (v4-7), matching A
      const v8bf* pb =
          (const v8bf*)&ly[(ni * 16 + ln) * LYP + kb * 32 + 8 * g];
      const v8bf b_lo = pb[0];
      const v8bf b_hi = pb[2];
      const v16bf bfm = __builtin_shufflevector(
          b_lo, b_hi, 0, 1, 2, 3, 4, 5, 6, 7, 8, 9, 10, 11, 12, 13, 14, 15);
      acc = __builtin_amdgcn_wmma_f32_16x16x32_bf16(
          false, afk[kb], false, bfm, (short)0, acc, false, false);
    }
    // D layout: lane n = ln, VGPR v -> row v + 8*g.  Replicate over s (12).
    #pragma unroll
    for (int v = 0; v < 8; ++v) {
      const int co = mi * 16 + v + 8 * g;
      const int p  = p0 + ni * 16 + ln;
      const float val = acc[v] + lb[co];
      float4 q = make_float4(val, val, val, val);
      float4* dst = (float4*)(out + (((size_t)b * NC + co) * (NH * NW) + p) * NS);
      dst[0] = q; dst[1] = q; dst[2] = q;
    }
  }
}

extern "C" void kernel_launch(void* const* d_in, const int* in_sizes, int n_in,
                              void* d_out, int out_size, void* d_ws, size_t ws_size,
                              hipStream_t stream) {
  const float* x  = (const float*)d_in[0];
  const float* w1 = (const float*)d_in[1];
  const float* b1 = (const float*)d_in[2];
  const float* w2 = (const float*)d_in[3];
  // d_in[4] (b2) cancels exactly in the softmax over t -> unused.

  char*   ws   = (char*)d_ws;
  float*  weff = (float*)(ws + 0);        // 25*64 f32   (6400 B)
  float*  wsum = (float*)(ws + 6400);     // 25 f32
  float*  b1v  = (float*)(ws + 6528);     // 64 f32
  __bf16* w1v  = (__bf16*)(ws + 6784);    // 64*64 bf16  (8192 B)
  float*  attn = (float*)(ws + 15360);    // B*H*W*S f32 (1.57 MB)
  float*  out  = (float*)d_out;

  hipLaunchKernelGGL(precompute_kernel, dim3(1), dim3(256), 0, stream,
                     w1, b1, w2, weff, wsum, b1v, w1v);
  hipLaunchKernelGGL(attn_kernel, dim3(NB * NH * 4), dim3(192), 0, stream,
                     x, weff, wsum, attn);
  hipLaunchKernelGGL(gemm_out_kernel, dim3(NB * 64), dim3(256), 0, stream,
                     x, attn, w1v, b1v, out);
}
